// AttentionAgger_15126874816598
// MI455X (gfx1250) — matrix operand order repaired and checked by reading, not compiled, not run on hardware
//
#include <hip/hip_runtime.h>
#include <hip/hip_bf16.h>

// ---------------------------------------------------------------------------
// AttentionAgger for MI455X (gfx1250, wave32, WMMA).
//   out = softmax( softmax(Q K^T / sqrt(512)) + softmax(mask/L) ) @ V
// B=2 H=8 L=2048 D=64.  Two-pass flash attention; all matmuls on
// v_wmma_f32_16x16x32_f16. S computed TRANSPOSED (A=K rows, B=Q^T) so the
// probability tile re-packs into the PV WMMA A-operand with per-lane register
// moves only (no LDS transpose). Each wave owns 32 queries (two 16-query
// tiles) so K/V fragment loads + f16 conversions are amortized over 2x WMMAs.
// Workspace: mT = softmax(mask/L) transposed (16 MB) + V^T (8 MB).
// ---------------------------------------------------------------------------

typedef __attribute__((ext_vector_type(16))) _Float16 v16h;
typedef __attribute__((ext_vector_type(8)))  float    v8f;

#define LSEQ 2048
#define DDIM 64
#define NBH  16

__device__ __forceinline__ v8f wmma16(v16h a, v16h b, v8f c) {
  return __builtin_amdgcn_wmma_f32_16x16x32_f16(false, a, false, b,
                                                (short)0, c, false, false);
}

__device__ __forceinline__ void cvt4(v16h& r, int base, float4 x, float s) {
  r[base + 0] = (_Float16)(x.x * s);
  r[base + 1] = (_Float16)(x.y * s);
  r[base + 2] = (_Float16)(x.z * s);
  r[base + 3] = (_Float16)(x.w * s);
}

// A-operand (16x32 f16): lane r=l&15 holds matrix row r.
// half h=l>>4: elems 0-7 = d0+h*8+0..7 ; elems 8-15 = d0+16+h*8+0..7
__device__ __forceinline__ v16h load_afrag(const float* row, int d0, int h) {
  const float4* p = (const float4*)(row + d0 + h * 8);
  const float4* q = (const float4*)(row + d0 + 16 + h * 8);
  v16h r;
  cvt4(r, 0, p[0], 1.0f);  cvt4(r, 4,  p[1], 1.0f);
  cvt4(r, 8, q[0], 1.0f);  cvt4(r, 12, q[1], 1.0f);
  return r;
}

// B-operand (32x16 f16): lane n=l&15 holds matrix column n;
// elems 0-7 = k = h*8+0..7 at p ; elems 8-15 = k = 16+h*8+0..7 at p+16.
__device__ __forceinline__ v16h load_bfrag16(const float* p, float s) {
  const float4* q = (const float4*)p;
  const float4* q2 = (const float4*)(p + 16);
  v16h r;
  cvt4(r, 0, q[0], s);  cvt4(r, 4,  q[1], s);
  cvt4(r, 8, q2[0], s);  cvt4(r, 12, q2[1], s);
  return r;
}

// ---- kernel 1: mT[j*L + i] = softmax_j(mask[i]/L)  (stored transposed) ----
__global__ __launch_bounds__(256)
void mask_softmax_T(const float* __restrict__ mask, float* __restrict__ mT) {
  __shared__ float red[256];
  const int i = blockIdx.x;
  const int t = threadIdx.x;
  const float inv = 1.0f / (float)LSEQ;
  float vals[8];
  float lmax = -1e30f;
#pragma unroll
  for (int k = 0; k < 8; k++) {
    float v = mask[(size_t)i * LSEQ + t + k * 256] * inv;
    vals[k] = v;
    lmax = fmaxf(lmax, v);
  }
  red[t] = lmax; __syncthreads();
  for (int s = 128; s > 0; s >>= 1) {
    if (t < s) red[t] = fmaxf(red[t], red[t + s]);
    __syncthreads();
  }
  const float rmax = red[0]; __syncthreads();
  float e[8], lsum = 0.0f;
#pragma unroll
  for (int k = 0; k < 8; k++) { e[k] = __expf(vals[k] - rmax); lsum += e[k]; }
  red[t] = lsum; __syncthreads();
  for (int s = 128; s > 0; s >>= 1) {
    if (t < s) red[t] += red[t + s];
    __syncthreads();
  }
  const float rs = 1.0f / red[0];
#pragma unroll
  for (int k = 0; k < 8; k++)
    mT[(size_t)(t + k * 256) * LSEQ + i] = e[k] * rs;
}

// ---- kernel 2: VT[bh][d][k] = V[bh][k][d] ----
__global__ __launch_bounds__(256)
void transpose_v(const float* __restrict__ V, float* __restrict__ VT) {
  const int idx = blockIdx.x * 256 + threadIdx.x;     // 16*2048*64 elements
  const int bh  = idx >> 17;
  const int rem = idx & 131071;
  const int k = rem >> 6, d = rem & 63;
  VT[(size_t)bh * (LSEQ * DDIM) + (size_t)d * LSEQ + k] = V[idx];
}

// ---- kernel 3: fused two-pass double-softmax attention ----
// One wave = 32 queries (two 16-query tiles u=0,1 sharing K/V fragments).
__global__ __launch_bounds__(128)
void attn_agger(const float* __restrict__ Q, const float* __restrict__ K,
                const float* __restrict__ VT, const float* __restrict__ mT,
                float* __restrict__ Out) {
  const int lane = threadIdx.x & 31;
  const int wave = threadIdx.x >> 5;
  const int h16  = lane >> 4;       // half-wave id
  const int c    = lane & 15;       // lane-in-half (query column)
  const int bh   = blockIdx.y;
  const int q0   = (blockIdx.x * 4 + wave) * 32;

  const size_t headOff = (size_t)bh * LSEQ * DDIM;
  const float* Qh  = Q  + headOff;
  const float* Kh  = K  + headOff;
  const float* VTh = VT + headOff;
  float*       Oh  = Out + headOff;

  const float scale = 0.04419417382415922f;  // 1/sqrt(512) (MDIM=512)

  // Q^T B-fragments: two 16-query tiles, scale folded in.
  v16h qb0[2], qb1[2];
#pragma unroll
  for (int u = 0; u < 2; u++) {
    const float* qrow = Qh + (size_t)(q0 + u * 16 + c) * DDIM;
    qb0[u] = load_bfrag16(qrow + 0  + h16 * 8, scale);
    qb1[u] = load_bfrag16(qrow + 32 + h16 * 8, scale);
  }

  // -------- pass 1: flash stats (per-lane: query=c, keys of this half) ----
  float M[2] = {-1e30f, -1e30f}, Z[2] = {0.0f, 0.0f};
  for (int kb = 0; kb < LSEQ; kb += 16) {
    const float* krow = Kh + (size_t)(kb + c) * DDIM;
    const v16h ka0 = load_afrag(krow, 0, h16);
    const v16h ka1 = load_afrag(krow, 32, h16);
#pragma unroll
    for (int u = 0; u < 2; u++) {
      v8f acc = {};
      acc = wmma16(ka0, qb0[u], acc);  // S^T tile: acc[v]=S[q][kb+v+8*h16]
      acc = wmma16(ka1, qb1[u], acc);
      float tmax = acc[0];
#pragma unroll
      for (int v = 1; v < 8; v++) tmax = fmaxf(tmax, acc[v]);
      const float nM = fmaxf(M[u], tmax);
      float s = 0.0f;
#pragma unroll
      for (int v = 0; v < 8; v++) s += __expf(acc[v] - nM);
      Z[u] = Z[u] * __expf(M[u] - nM) + s;
      M[u] = nM;
    }
  }
  float invZ[2];
#pragma unroll
  for (int u = 0; u < 2; u++) {  // merge the two halves' partial stats
    const float Mo = __shfl_xor(M[u], 16, 32);
    const float Zo = __shfl_xor(Z[u], 16, 32);
    const float nM = fmaxf(M[u], Mo);
    Z[u] = Z[u] * __expf(M[u] - nM) + Zo * __expf(Mo - nM);
    M[u] = nM;
    invZ[u] = 1.0f / Z[u];
  }

  // -------- pass 2: t = exp(softmax1 + m); O += t*V; s2 = sum t ----------
  float s2[2] = {0.0f, 0.0f};
  v8f o[2][4];
#pragma unroll
  for (int u = 0; u < 2; u++)
#pragma unroll
    for (int n = 0; n < 4; n++) o[u][n] = (v8f){};

  for (int kb = 0; kb < LSEQ; kb += 32) {
    v16h ap[2];  // P A-fragments (16x32) for the two query tiles
#pragma unroll
    for (int tt = 0; tt < 2; tt++) {
      const int kbase = kb + 16 * tt;
      const float* krow = Kh + (size_t)(kbase + c) * DDIM;
      const v16h ka0 = load_afrag(krow, 0, h16);
      const v16h ka1 = load_afrag(krow, 32, h16);
#pragma unroll
      for (int u = 0; u < 2; u++) {
        v8f acc = {};
        acc = wmma16(ka0, qb0[u], acc);
        acc = wmma16(ka1, qb1[u], acc);
        const float* mcol =
            mT + (size_t)(kbase + 8 * h16) * LSEQ + q0 + u * 16 + c;
#pragma unroll
        for (int v = 0; v < 8; v++) {
          const float p  = __expf(acc[v] - M[u]) * invZ[u];  // softmax1 prob
          const float mv = mcol[(size_t)v * LSEQ];           // m[q][key]
          const float tv = __expf(p + mv);                   // p+mv in [0,2]
          s2[u] += tv;
          if (tt == 0) ap[u][v] = (_Float16)tv;
          else         ap[u][8 + v] = (_Float16)tv;
        }
      }
    }
    // O(16q x 64d) += P(16q x 32k) * V(32k x 64d); V^T fragment shared by u.
    const float* vtb = VTh + kb + h16 * 8;
#pragma unroll
    for (int n = 0; n < 4; n++) {
      const v16h vb = load_bfrag16(vtb + (size_t)(c + 16 * n) * LSEQ, 1.0f);
      o[0][n] = wmma16(ap[0], vb, o[0][n]);
      o[1][n] = wmma16(ap[1], vb, o[1][n]);
    }
  }

  // Epilogue: divide by second-softmax denominator and store.
#pragma unroll
  for (int u = 0; u < 2; u++) {
    const float s2f = s2[u] + __shfl_xor(s2[u], 16, 32);
#pragma unroll
    for (int v = 0; v < 8; v++) {
      const float r = 1.0f / __shfl(s2f, v + 8 * h16, 32);
      float* orow = Oh + (size_t)(q0 + u * 16 + v + 8 * h16) * DDIM + c;
      orow[0]  = o[u][0][v] * r;
      orow[16] = o[u][1][v] * r;
      orow[32] = o[u][2][v] * r;
      orow[48] = o[u][3][v] * r;
    }
  }
}

// ---------------------------------------------------------------------------
extern "C" void kernel_launch(void* const* d_in, const int* in_sizes, int n_in,
                              void* d_out, int out_size, void* d_ws, size_t ws_size,
                              hipStream_t stream) {
  (void)in_sizes; (void)n_in; (void)out_size; (void)ws_size;
  const float* Q    = (const float*)d_in[0];
  const float* K    = (const float*)d_in[1];
  const float* V    = (const float*)d_in[2];
  const float* mask = (const float*)d_in[3];
  float* out = (float*)d_out;

  float* mT = (float*)d_ws;                                     // 16 MB
  float* VT = (float*)((char*)d_ws + (size_t)LSEQ * LSEQ * 4);  // +8 MB

  mask_softmax_T<<<LSEQ, 256, 0, stream>>>(mask, mT);
  transpose_v<<<(NBH * LSEQ * DDIM) / 256, 256, 0, stream>>>(V, VT);
  // 32 queries per wave, 4 waves per block: grid.x = 2048 / 128
  attn_agger<<<dim3(LSEQ / 128, NBH), 128, 0, stream>>>(Q, K, VT, mT, out);
}
